// DeepSeekWithEmbedding_65025804861467
// MI455X (gfx1250) — matrix operand order repaired and checked by reading, not compiled, not run on hardware
//
#include <hip/hip_runtime.h>
#include <hip/hip_bf16.h>

// ---------------------------------------------------------------------------
// Types for CDNA5 WMMA
// ---------------------------------------------------------------------------
typedef __attribute__((ext_vector_type(16))) __bf16 v16bf;
typedef __attribute__((ext_vector_type(8)))  float  v8f;

union Frag16 {            // 16 bf16 = 8 VGPRs = two 16-byte chunks
    uint4  q[2];
    v16bf  v;
};

// Native float->bf16 conversion: backend picks the hardware cvt path.
__device__ __forceinline__ unsigned short bf16u(float a) {
    __bf16 h = (__bf16)a;
    unsigned short u; __builtin_memcpy(&u, &h, 2); return u;
}
__device__ __forceinline__ unsigned int pack2(float a, float b) {
    return (unsigned int)bf16u(a) | ((unsigned int)bf16u(b) << 16);
}

// ---------------------------------------------------------------------------
// CDNA5 async global->LDS copy (ASYNCcnt-tracked DMA-style load) + wait
// ---------------------------------------------------------------------------
__device__ __forceinline__ void async_ld_b128(const void* gptr, void* lptr) {
    unsigned lds = (unsigned)(unsigned long long)lptr;   // low 32 bits = LDS byte addr
    asm volatile("global_load_async_to_lds_b128 %0, %1, off"
                 :: "v"(lds), "v"(gptr)
                 : "memory");
}
__device__ __forceinline__ void wait_async0() {
    asm volatile("s_wait_asynccnt 0x0" ::: "memory");
}

// ---------------------------------------------------------------------------
// Generic WMMA GEMM (fp32 in / fp32 out, bf16 compute):
//   C[z] = alpha * A[z] x (B[z] or B[z]^T) (+bias)(+residual)(+causal mask)
// Block: 256 threads (8 waves). Block tile 128x128, K-step 32, double-buffered
// LDS with in-register fp32->bf16 conversion while staging. 8 wmma/wave/K-step.
// Requires: M%128==0, N%128==0, K%32==0.
// ---------------------------------------------------------------------------
template <bool TRANS_B>
__global__ __launch_bounds__(256)
void gemm_bf16_wmma(const float* __restrict__ A, const float* __restrict__ B,
                    float* __restrict__ C,
                    const float* __restrict__ bias,
                    const float* __restrict__ residual,
                    int M, int N, int K, int lda, int ldb, int ldc,
                    float alpha, int zdiv,
                    long long sa1, long long sa2,
                    long long sb1, long long sb2,
                    long long sc1, long long sc2,
                    int causal)
{
    constexpr int BM = 128, BN = 128, BK = 32;
    constexpr int LDH = BK + 8;     // 40 bf16 (80B) row stride: conflict-free, 16B aligned

    __shared__ __align__(16) unsigned short sA[2][BM * LDH];  // 2 x 10240 B
    __shared__ __align__(16) unsigned short sB[2][BN * LDH];  // 2 x 10240 B

    const int tid  = threadIdx.x;
    const int lane = tid & 31;
    const int wave = tid >> 5;
    const int l15  = lane & 15;
    const int lhi  = lane >> 4;           // 0 or 1 (half-wave)
    const int wm   = (wave & 3) * 32;     // wave M offset in block tile
    const int wn   = (wave >> 2) * 64;    // wave N offset in block tile

    const int z  = blockIdx.z;
    const int z1 = z / zdiv;
    const int z2 = z - z1 * zdiv;

    const long long cbase = z1 * sc1 + z2 * sc2;
    float* Cz = C + cbase;

    // ---- causal early-out: whole tile strictly above the diagonal ----------
    if (causal && blockIdx.x > blockIdx.y) {
        const float4 mv = make_float4(-3.4028234663852886e38f, -3.4028234663852886e38f,
                                      -3.4028234663852886e38f, -3.4028234663852886e38f);
        #pragma unroll
        for (int i = 0; i < 16; ++i) {                 // 4096 float4 / 256 threads
            int lin = tid + i * 256;
            int r   = lin >> 5;
            int c   = (lin & 31) << 2;
            *reinterpret_cast<float4*>(
                &Cz[(long long)(blockIdx.y * BM + r) * ldc + blockIdx.x * BN + c]) = mv;
        }
        return;
    }

    const float* Ab = A + z1 * sa1 + z2 * sa2 + (long long)blockIdx.y * BM * lda;
    const float* Bz = B + z1 * sb1 + z2 * sb2;

    // ---- tile staging (fp32 global -> bf16 LDS) ----------------------------
    auto stageA = [&](unsigned short* dst, int kb, bool pf) {
        #pragma unroll
        for (int i = 0; i < 4; ++i) {
            int lin = tid + i * 256;                   // 1024 float4 units (128x32)
            int r   = lin >> 3;                        // row 0..127
            int c   = (lin & 7) << 2;                  // k 0..28 step 4
            const float* ap = Ab + (long long)r * lda + kb + c;
            float4 f = *reinterpret_cast<const float4*>(ap);
            if (pf) __builtin_prefetch(ap + BK, 0, 1); // global_prefetch_b8 (next tile)
            *reinterpret_cast<unsigned int*>(&dst[r * LDH + c])     = pack2(f.x, f.y);
            *reinterpret_cast<unsigned int*>(&dst[r * LDH + c + 2]) = pack2(f.z, f.w);
        }
    };
    auto stageB = [&](unsigned short* dst, int kb, bool pf) {
        if (TRANS_B) {
            // B stored [N,K]: direct copy rows n0..n0+127
            const float* Bb = Bz + (long long)blockIdx.x * BN * ldb;
            #pragma unroll
            for (int i = 0; i < 4; ++i) {
                int lin = tid + i * 256;
                int r   = lin >> 3;                    // n 0..127
                int c   = (lin & 7) << 2;              // k
                const float* bp = Bb + (long long)r * ldb + kb + c;
                float4 f = *reinterpret_cast<const float4*>(bp);
                if (pf) __builtin_prefetch(bp + BK, 0, 1);
                *reinterpret_cast<unsigned int*>(&dst[r * LDH + c])     = pack2(f.x, f.y);
                *reinterpret_cast<unsigned int*>(&dst[r * LDH + c + 2]) = pack2(f.z, f.w);
            }
        } else {
            // B stored [K,N]: transpose while staging -> LDS[n][k]
            const float* Bb = Bz + (long long)blockIdx.x * BN;
            #pragma unroll
            for (int i = 0; i < 4; ++i) {
                int lin = tid + i * 256;               // 1024 float4 units (32k x 32n4)
                int r   = lin >> 5;                    // k 0..31
                int c   = (lin & 31) << 2;             // n 0..124 step 4
                const float* bp = Bb + (long long)(kb + r) * ldb + c;
                float4 f = *reinterpret_cast<const float4*>(bp);
                if (pf) __builtin_prefetch(bp + (long long)BK * ldb, 0, 1);
                dst[(c + 0) * LDH + r] = bf16u(f.x);
                dst[(c + 1) * LDH + r] = bf16u(f.y);
                dst[(c + 2) * LDH + r] = bf16u(f.z);
                dst[(c + 3) * LDH + r] = bf16u(f.w);
            }
        }
    };

    v8f acc[2][4] = {};

    stageA(sA[0], 0, BK < K);
    stageB(sB[0], 0, BK < K);

    int buf = 0;
    for (int kb = 0; kb < K; kb += BK) {
        __syncthreads();                               // staged tile `buf` is ready
        int nkb = kb + BK;
        if (nkb < K) {                                 // stage next tile into other buffer
            bool pf = (nkb + BK) < K;
            stageA(sA[buf ^ 1], nkb, pf);
            stageB(sB[buf ^ 1], nkb, pf);
        }

        // ---- fragments (ISA 16-bit A 16x32 / B 32x16 wave32 layouts) -------
        const unsigned short* cA = sA[buf];
        const unsigned short* cB = sB[buf];
        Frag16 afr[2], bfr[4];
        #pragma unroll
        for (int mi = 0; mi < 2; ++mi) {
            int m  = wm + mi * 16 + l15;
            int ko = lhi * 8;
            afr[mi].q[0] = *reinterpret_cast<const uint4*>(&cA[m * LDH + ko]);
            afr[mi].q[1] = *reinterpret_cast<const uint4*>(&cA[m * LDH + ko + 16]);
        }
        #pragma unroll
        for (int ni = 0; ni < 4; ++ni) {
            int n  = wn + ni * 16 + l15;
            int kc = lhi * 8;
            bfr[ni].q[0] = *reinterpret_cast<const uint4*>(&cB[n * LDH + kc]);
            bfr[ni].q[1] = *reinterpret_cast<const uint4*>(&cB[n * LDH + kc + 16]);
        }

        #pragma unroll
        for (int mi = 0; mi < 2; ++mi)
            #pragma unroll
            for (int ni = 0; ni < 4; ++ni)
                acc[mi][ni] = __builtin_amdgcn_wmma_f32_16x16x32_bf16(
                    false, afr[mi].v, false, bfr[ni].v,
                    (short)0, acc[mi][ni], false, false);

        buf ^= 1;
    }

    // ---- epilogue ----------------------------------------------------------
    const float* Rz = residual ? residual + cbase : nullptr;
    #pragma unroll
    for (int mi = 0; mi < 2; ++mi) {
        #pragma unroll
        for (int ni = 0; ni < 4; ++ni) {
            int gmb = blockIdx.y * BM + wm + mi * 16 + lhi * 8;
            int gn  = blockIdx.x * BN + wn + ni * 16 + l15;
            #pragma unroll
            for (int r = 0; r < 8; ++r) {
                int row = gmb + r;
                float v = alpha * acc[mi][ni][r];
                if (bias)     v += bias[gn];
                if (Rz)       v += Rz[(long long)row * ldc + gn];
                if (causal && gn > row) v = -3.4028234663852886e38f;
                Cz[(long long)row * ldc + gn] = v;
            }
        }
    }
}

// ---------------------------------------------------------------------------
// Attention WMMA GEMM (bf16 in / fp32 out), async-to-LDS staged:
//   C[z] = alpha * A[z] x B[z]^T  (A:[M,K] bf16, B:[N,K] bf16 row-major)
// Tiles are already bf16 in memory, so staging is a raw byte copy: use
// GLOBAL_LOAD_ASYNC_TO_LDS_B128 (ASYNCcnt) — the async engine fills tile k+1
// while the waves run WMMAs on tile k. Block tile 128x128, K-step 64 bf16.
// Requires: M%128==0, N%128==0, K%64==0, rows 16B-aligned.
// ---------------------------------------------------------------------------
__global__ __launch_bounds__(256)
void attn_gemm_async(const unsigned short* __restrict__ A,
                     const unsigned short* __restrict__ B,
                     float* __restrict__ C,
                     int M, int N, int K, int lda, int ldb, int ldc,
                     float alpha, long long saz, long long sbz,
                     int zdiv, long long sc1, long long sc2, int causal)
{
    constexpr int BM = 128, BN = 128, BK = 64;
    constexpr int LDH = BK + 8;     // 72 bf16 rows (144B): 16B aligned, conflict-spread

    __shared__ __align__(16) unsigned short sA[2][BM * LDH];  // 2 x 18432 B
    __shared__ __align__(16) unsigned short sB[2][BN * LDH];  // 2 x 18432 B

    const int tid  = threadIdx.x;
    const int lane = tid & 31;
    const int wave = tid >> 5;
    const int l15  = lane & 15;
    const int lhi  = lane >> 4;
    const int wm   = (wave & 3) * 32;
    const int wn   = (wave >> 2) * 64;

    const int z  = blockIdx.z;
    const int z1 = z / zdiv;
    const int z2 = z - z1 * zdiv;
    float* Cz = C + z1 * sc1 + z2 * sc2;

    if (causal && blockIdx.x > blockIdx.y) {
        const float4 mv = make_float4(-3.4028234663852886e38f, -3.4028234663852886e38f,
                                      -3.4028234663852886e38f, -3.4028234663852886e38f);
        #pragma unroll
        for (int i = 0; i < 16; ++i) {
            int lin = tid + i * 256;
            int r   = lin >> 5;
            int c   = (lin & 31) << 2;
            *reinterpret_cast<float4*>(
                &Cz[(long long)(blockIdx.y * BM + r) * ldc + blockIdx.x * BN + c]) = mv;
        }
        return;
    }

    const unsigned short* Ab = A + (long long)z * saz + (long long)blockIdx.y * BM * lda;
    const unsigned short* Bb = B + (long long)z * sbz + (long long)blockIdx.x * BN * ldb;

    // stage one 128x64 bf16 tile of A and of B: 8 async b128 per thread
    auto stage = [&](int buf, int kb) {
        #pragma unroll
        for (int i = 0; i < 4; ++i) {
            int lin = tid + i * 256;                   // 1024 x 16B chunks per matrix
            int r   = lin >> 3;                        // row 0..127
            int c   = (lin & 7) << 3;                  // bf16 col 0..56 step 8 (16B)
            async_ld_b128(Ab + (long long)r * lda + kb + c, &sA[buf][r * LDH + c]);
            async_ld_b128(Bb + (long long)r * ldb + kb + c, &sB[buf][r * LDH + c]);
        }
    };

    v8f acc[2][4] = {};
    stage(0, 0);

    int buf = 0;
    for (int kb = 0; kb < K; kb += BK) {
        wait_async0();                                 // my async copies into `buf` done
        __syncthreads();                               // everyone's copies visible
        if (kb + BK < K) stage(buf ^ 1, kb + BK);      // async engine fills next tile

        const unsigned short* cA = sA[buf];
        const unsigned short* cB = sB[buf];
        #pragma unroll
        for (int kk = 0; kk < BK; kk += 32) {
            Frag16 afr[2], bfr[4];
            #pragma unroll
            for (int mi = 0; mi < 2; ++mi) {
                int m = wm + mi * 16 + l15;
                int ko = kk + lhi * 8;
                afr[mi].q[0] = *reinterpret_cast<const uint4*>(&cA[m * LDH + ko]);
                afr[mi].q[1] = *reinterpret_cast<const uint4*>(&cA[m * LDH + ko + 16]);
            }
            #pragma unroll
            for (int ni = 0; ni < 4; ++ni) {
                int n = wn + ni * 16 + l15;
                int kc = kk + lhi * 8;
                bfr[ni].q[0] = *reinterpret_cast<const uint4*>(&cB[n * LDH + kc]);
                bfr[ni].q[1] = *reinterpret_cast<const uint4*>(&cB[n * LDH + kc + 16]);
            }
            #pragma unroll
            for (int mi = 0; mi < 2; ++mi)
                #pragma unroll
                for (int ni = 0; ni < 4; ++ni)
                    acc[mi][ni] = __builtin_amdgcn_wmma_f32_16x16x32_bf16(
                        false, afr[mi].v, false, bfr[ni].v,
                        (short)0, acc[mi][ni], false, false);
        }
        buf ^= 1;
    }

    #pragma unroll
    for (int mi = 0; mi < 2; ++mi) {
        #pragma unroll
        for (int ni = 0; ni < 4; ++ni) {
            int gmb = blockIdx.y * BM + wm + mi * 16 + lhi * 8;
            int gn  = blockIdx.x * BN + wn + ni * 16 + l15;
            #pragma unroll
            for (int r = 0; r < 8; ++r) {
                int row = gmb + r;
                float v = alpha * acc[mi][ni][r];
                if (causal && gn > row) v = -3.4028234663852886e38f;
                Cz[(long long)row * ldc + gn] = v;
            }
        }
    }
}

// ---------------------------------------------------------------------------
// RMSNorm: one block (256 threads) per row
// ---------------------------------------------------------------------------
__global__ __launch_bounds__(256)
void rmsnorm_kernel(const float* __restrict__ x, const float* __restrict__ w,
                    float* __restrict__ out, int ncols)
{
    __shared__ float red[256];
    const long long row = blockIdx.x;
    const float* xr = x + row * ncols;
    float ss = 0.f;
    for (int c = threadIdx.x; c < ncols; c += 256) { float v = xr[c]; ss += v * v; }
    red[threadIdx.x] = ss; __syncthreads();
    for (int s = 128; s > 0; s >>= 1) {
        if (threadIdx.x < s) red[threadIdx.x] += red[threadIdx.x + s];
        __syncthreads();
    }
    float scale = rsqrtf(red[0] / (float)ncols + 1e-6f);
    for (int c = threadIdx.x; c < ncols; c += 256)
        out[row * ncols + c] = xr[c] * scale * w[c];
}

// ---------------------------------------------------------------------------
// RoPE: read fp32 q/k [B,S,NH,HD], write rotated bf16 head-major [B,NH,S,HD]
// ---------------------------------------------------------------------------
__global__ void rope_bf16_kernel(const float* __restrict__ t,
                                 unsigned short* __restrict__ th, int total_half)
{
    int i = blockIdx.x * blockDim.x + threadIdx.x;
    if (i >= total_half) return;
    int j    = i & 63;                 // freq index 0..63 (HD/2)
    int rest = i >> 6;                 // (b*S + s)*NH + h
    int hh   = rest & 31;              // NH=32
    int bs   = rest >> 5;              // b*S + s
    int s    = bs & 511;               // S=512
    int b    = bs >> 9;
    long long ibase = ((long long)rest << 7) + j;                       // [B,S,NH,HD]
    long long obase = ((((long long)(b * 32 + hh)) * 512 + s) << 7) + j; // [B,NH,S,HD]
    float lo = t[ibase], hi = t[ibase + 64];
    float freq = __expf(-(float)j * 0.14391156831212787f);  // ln(10000)/64
    float sn, cs; __sincosf((float)s * freq, &sn, &cs);
    th[obase]      = bf16u(lo * cs - hi * sn);
    th[obase + 64] = bf16u(hi * cs + lo * sn);
}

// ---------------------------------------------------------------------------
// V: fp32 [B,S,NH,HD] -> bf16 transposed head-major [B,NH,HD,S]
// ---------------------------------------------------------------------------
__global__ void v_bf16t_kernel(const float* __restrict__ v,
                               unsigned short* __restrict__ vt, int total)
{
    int i = blockIdx.x * blockDim.x + threadIdx.x;
    if (i >= total) return;
    int s  = i & 511;                  // S=512
    int d  = (i >> 9) & 127;           // HD=128
    int hh = (i >> 16) & 31;           // NH=32
    int b  = i >> 21;
    vt[i] = bf16u(v[((((long long)(b * 512 + s)) * 32 + hh) << 7) + d]);
}

// ---------------------------------------------------------------------------
// Row softmax: fp32 scores in, bf16 probabilities out. One block per row.
// ---------------------------------------------------------------------------
__global__ __launch_bounds__(256)
void softmax_bf16_kernel(float* __restrict__ s, unsigned short* __restrict__ pb, int n)
{
    __shared__ float red[256];
    float* p = s + (long long)blockIdx.x * n;
    unsigned short* pr = pb + (long long)blockIdx.x * n;
    float m = -3.4028234663852886e38f;
    for (int c = threadIdx.x; c < n; c += 256) m = fmaxf(m, p[c]);
    red[threadIdx.x] = m; __syncthreads();
    for (int st = 128; st > 0; st >>= 1) {
        if (threadIdx.x < st) red[threadIdx.x] = fmaxf(red[threadIdx.x], red[threadIdx.x + st]);
        __syncthreads();
    }
    float mx = red[0]; __syncthreads();
    float sum = 0.f;
    for (int c = threadIdx.x; c < n; c += 256) { float e = __expf(p[c] - mx); p[c] = e; sum += e; }
    red[threadIdx.x] = sum; __syncthreads();
    for (int st = 128; st > 0; st >>= 1) {
        if (threadIdx.x < st) red[threadIdx.x] += red[threadIdx.x + st];
        __syncthreads();
    }
    float inv = 1.f / red[0];
    for (int c = threadIdx.x; c < n; c += 256) pr[c] = bf16u(p[c] * inv);
}

// ---------------------------------------------------------------------------
// g = silu(g) * u, elementwise
// ---------------------------------------------------------------------------
__global__ void silu_mul_kernel(float* __restrict__ g, const float* __restrict__ u,
                                long long n)
{
    long long i = (long long)blockIdx.x * blockDim.x + threadIdx.x;
    if (i < n) { float x = g[i]; g[i] = (x / (1.f + __expf(-x))) * u[i]; }
}

// ---------------------------------------------------------------------------
// Final tiny projection: [rows,4096] x [4096,8] + b. One wave per output.
// ---------------------------------------------------------------------------
__global__ __launch_bounds__(256)
void outproj_kernel(const float* __restrict__ hn, const float* __restrict__ W,
                    const float* __restrict__ b, float* __restrict__ out)
{
    int row  = blockIdx.x;
    int col  = threadIdx.x >> 5;
    int lane = threadIdx.x & 31;
    const float* hr = hn + (long long)row * 4096;
    float sum = 0.f;
    for (int k = lane; k < 4096; k += 32) sum += hr[k] * W[(long long)k * 8 + col];
    for (int o = 16; o > 0; o >>= 1) sum += __shfl_xor(sum, o, 32);
    if (lane == 0) out[row * 8 + col] = sum + b[col];
}

// ---------------------------------------------------------------------------
// Host-side dispatch
// ---------------------------------------------------------------------------
static void gemm(hipStream_t st, const float* A, const float* B, float* C,
                 const float* bias, const float* residual,
                 int M, int N, int K, int lda, int ldb, int ldc,
                 float alpha, int batch, int zdiv,
                 long long sa1, long long sa2, long long sb1, long long sb2,
                 long long sc1, long long sc2, bool transB, bool causal)
{
    dim3 grid(N / 128, M / 128, batch), block(256);
    if (transB)
        gemm_bf16_wmma<true><<<grid, block, 0, st>>>(A, B, C, bias, residual,
            M, N, K, lda, ldb, ldc, alpha, zdiv, sa1, sa2, sb1, sb2, sc1, sc2,
            causal ? 1 : 0);
    else
        gemm_bf16_wmma<false><<<grid, block, 0, st>>>(A, B, C, bias, residual,
            M, N, K, lda, ldb, ldc, alpha, zdiv, sa1, sa2, sb1, sb2, sc1, sc2,
            causal ? 1 : 0);
}

extern "C" void kernel_launch(void* const* d_in, const int* in_sizes, int n_in,
                              void* d_out, int out_size, void* d_ws, size_t ws_size,
                              hipStream_t stream)
{
    constexpr int L = 2, H = 4096, NH = 32, HD = 128, F = 11008;
    constexpr int B = 2, S = 512, NIN = 512;
    constexpr int BS = B * S;                   // 1024 rows
    const float scale = 0.08838834764831845f;   // 1/sqrt(HD)

    const float* x     = (const float*)d_in[0];
    const float* W_emb = (const float*)d_in[1];
    const float* b_emb = (const float*)d_in[2];
    const float* ln1_w = (const float*)d_in[3];
    const float* Wq    = (const float*)d_in[4];
    const float* Wk    = (const float*)d_in[5];
    const float* Wv    = (const float*)d_in[6];
    const float* Wo    = (const float*)d_in[7];
    const float* ln2_w = (const float*)d_in[8];
    const float* Wg    = (const float*)d_in[9];
    const float* Wu    = (const float*)d_in[10];
    const float* Wd    = (const float*)d_in[11];
    const float* lnf_w = (const float*)d_in[12];
    const float* W_out = (const float*)d_in[13];
    const float* b_out = (const float*)d_in[14];
    float* out = (float*)d_out;

    // workspace carve-out: fp32 region then bf16 region (all 16B aligned)
    float* p = (float*)d_ws;
    float* h      = p; p += (size_t)BS * H;
    float* hn     = p; p += (size_t)BS * H;
    float* q      = p; p += (size_t)BS * H;
    float* k      = p; p += (size_t)BS * H;
    float* v      = p; p += (size_t)BS * H;
    float* o      = p; p += (size_t)BS * H;
    float* scores = p; p += (size_t)B * NH * S * S;
    float* g      = p; p += (size_t)BS * F;
    float* u      = p; p += (size_t)BS * F;
    unsigned short* us = (unsigned short*)p;
    unsigned short* qh  = us; us += (size_t)B * NH * S * HD;   // bf16 [B,NH,S,HD]
    unsigned short* kh  = us; us += (size_t)B * NH * S * HD;   // bf16 [B,NH,S,HD]
    unsigned short* vt  = us; us += (size_t)B * NH * HD * S;   // bf16 [B,NH,HD,S]
    unsigned short* pbf = us; us += (size_t)B * NH * S * S;    // bf16 [z,S,S]

    const long long sSH = (long long)S * H;
    const long long sSS = (long long)S * S;

    // embedding: h = x @ W_emb + b_emb
    gemm(stream, x, W_emb, h, b_emb, nullptr, BS, H, NIN, NIN, H, H,
         1.f, 1, 1, 0, 0, 0, 0, 0, 0, false, false);

    for (int l = 0; l < L; ++l) {
        const float* Wql = Wq + (size_t)l * H * H;
        const float* Wkl = Wk + (size_t)l * H * H;
        const float* Wvl = Wv + (size_t)l * H * H;
        const float* Wol = Wo + (size_t)l * H * H;
        const float* Wgl = Wg + (size_t)l * H * F;
        const float* Wul = Wu + (size_t)l * H * F;
        const float* Wdl = Wd + (size_t)l * F * H;

        // attention
        rmsnorm_kernel<<<BS, 256, 0, stream>>>(h, ln1_w + (size_t)l * H, hn, H);
        gemm(stream, hn, Wql, q, nullptr, nullptr, BS, H, H, H, H, H,
             1.f, 1, 1, 0, 0, 0, 0, 0, 0, false, false);
        gemm(stream, hn, Wkl, k, nullptr, nullptr, BS, H, H, H, H, H,
             1.f, 1, 1, 0, 0, 0, 0, 0, 0, false, false);
        gemm(stream, hn, Wvl, v, nullptr, nullptr, BS, H, H, H, H, H,
             1.f, 1, 1, 0, 0, 0, 0, 0, 0, false, false);

        // RoPE + pack to bf16 head-major; V -> bf16 transposed head-major
        int half = BS * H / 2;
        rope_bf16_kernel<<<(half + 255) / 256, 256, 0, stream>>>(q, qh, half);
        rope_bf16_kernel<<<(half + 255) / 256, 256, 0, stream>>>(k, kh, half);
        int tot = BS * H;
        v_bf16t_kernel<<<(tot + 255) / 256, 256, 0, stream>>>(v, vt, tot);

        // scores[z] = scale * Q_z @ K_z^T (+causal); async-to-LDS bf16 staging
        attn_gemm_async<<<dim3(S / 128, S / 128, B * NH), 256, 0, stream>>>(
            qh, kh, scores, S, S, HD, HD, HD, S,
            scale, (long long)S * HD, (long long)S * HD,
            1, sSS, 0, 1);

        softmax_bf16_kernel<<<B * NH * S, 256, 0, stream>>>(scores, pbf, S);

        // o[z] = P_z @ V_z  (B operand pre-transposed: vt rows are n, cols k)
        attn_gemm_async<<<dim3(HD / 128, S / 128, B * NH), 256, 0, stream>>>(
            pbf, vt, o, S, HD, S, S, S, H,
            1.f, sSS, (long long)HD * S,
            NH, sSH, HD, 0);

        // h = h + o @ Wo
        gemm(stream, o, Wol, h, nullptr, h, BS, H, H, H, H, H,
             1.f, 1, 1, 0, 0, 0, 0, 0, 0, false, false);

        // MLP
        rmsnorm_kernel<<<BS, 256, 0, stream>>>(h, ln2_w + (size_t)l * H, hn, H);
        gemm(stream, hn, Wgl, g, nullptr, nullptr, BS, F, H, H, F, F,
             1.f, 1, 1, 0, 0, 0, 0, 0, 0, false, false);
        gemm(stream, hn, Wul, u, nullptr, nullptr, BS, F, H, H, F, F,
             1.f, 1, 1, 0, 0, 0, 0, 0, 0, false, false);
        long long nf = (long long)BS * F;
        silu_mul_kernel<<<(unsigned)((nf + 255) / 256), 256, 0, stream>>>(g, u, nf);
        gemm(stream, g, Wdl, h, nullptr, h, BS, H, F, F, H, H,
             1.f, 1, 1, 0, 0, 0, 0, 0, 0, false, false);
    }

    // final norm + tiny projection
    rmsnorm_kernel<<<BS, 256, 0, stream>>>(h, lnf_w, hn, H);
    outproj_kernel<<<BS, 256, 0, stream>>>(hn, W_out, b_out, out);
}
